// WidthCharTransformerMLPHeaderFast_63754494542659
// MI455X (gfx1250) — matrix operand order repaired and checked by reading, not compiled, not run on hardware
//
#include <hip/hip_runtime.h>
#include <hip/hip_bf16.h>

// V=512, H=8, D=64, L=16, B=4096, F1=64, F2=64, F3=8
typedef __attribute__((ext_vector_type(16))) _Float16 v16h;
typedef __attribute__((ext_vector_type(8)))  float    v8f;

__device__ __forceinline__ v8f wmma16(v16h a, v16h b, v8f c) {
  return __builtin_amdgcn_wmma_f32_16x16x32_f16(
      /*neg_a=*/false, a, /*neg_b=*/false, b,
      /*c_mod=*/(short)0, c, /*reuse_a=*/false, /*reuse_b=*/false);
}

// Packed weight blob layout (bytes), total 16384:
//   0     : w1T [64][32] f16  (rows k>=8 zero)           4096 B
//   4096  : w2T [64][64] f16                             8192 B
//   12288 : w3T [16][64] f16  (cols n>=8 zero)           2048 B
//   14336 : f32 consts: bias1[64] bias2[64] bias3[16] gam[64] bet[64]
//   15424 : pad to 16384
#define BLOB_BYTES 16384

// ---------------------------------------------------------------------------
// Kernel 0: pack weights once into f16 WMMA-ready layout.
// ---------------------------------------------------------------------------
__global__ void k_prep(const float* __restrict__ w1, const float* __restrict__ w2,
                       const float* __restrict__ w3, const float* __restrict__ b1,
                       const float* __restrict__ b2, const float* __restrict__ b3,
                       const float* __restrict__ gamma, const float* __restrict__ beta,
                       unsigned char* __restrict__ blob) {
  int tid = threadIdx.x;  // single block, 256 threads
  _Float16* w1T = (_Float16*)(blob);
  _Float16* w2T = (_Float16*)(blob + 4096);
  _Float16* w3T = (_Float16*)(blob + 12288);
  float*    fb  = (float*)(blob + 14336);
  for (int i = tid; i < 64 * 32; i += 256) {
    int n = i >> 5, kk = i & 31;
    w1T[i] = (kk < 8) ? (_Float16)w1[kk * 64 + n] : (_Float16)0.f;
  }
  for (int i = tid; i < 64 * 64; i += 256) {
    int n = i >> 6, kk = i & 63;
    w2T[i] = (_Float16)w2[kk * 64 + n];
  }
  for (int i = tid; i < 16 * 64; i += 256) {
    int n = i >> 6, kk = i & 63;
    w3T[i] = (n < 8) ? (_Float16)w3[kk * 8 + n] : (_Float16)0.f;
  }
  if (tid < 64) fb[tid] = b1[tid];
  if (tid < 64) fb[64 + tid] = b2[tid];
  if (tid < 16) fb[128 + tid] = (tid < 8) ? b3[tid] : 0.f;
  if (tid < 64) fb[144 + tid] = gamma[tid];
  if (tid < 64) fb[208 + tid] = beta[tid];
  // pad region: fill deterministically
  for (int i = 15424 / 4 + tid; i < BLOB_BYTES / 4; i += 256) ((int*)blob)[i] = 0;
}

// ---------------------------------------------------------------------------
// Kernel 1: qkv = emb0 @ w_qkv  (512x64 @ 64x1536), split into q/k/v f16.
// ---------------------------------------------------------------------------
__global__ void k_qkv(const float* __restrict__ emb, const float* __restrict__ w_qkv,
                      _Float16* __restrict__ q, _Float16* __restrict__ k,
                      _Float16* __restrict__ v) {
  int idx = blockIdx.x * blockDim.x + threadIdx.x;
  if (idx >= 512 * 1536) return;
  int row = idx / 1536;
  int col = idx - row * 1536;
  float acc = 0.f;
  if (row != 0) {  // emb0.at[0].set(0)
    for (int t = 0; t < 64; ++t) acc += emb[row * 64 + t] * w_qkv[t * 1536 + col];
  }
  int grp = col >> 9;       // 0:q 1:k 2:v
  int within = col & 511;   // h*64 + d
  _Float16 hv = (_Float16)acc;
  if (grp == 0)      q[row * 512 + within] = hv;
  else if (grp == 1) k[row * 512 + within] = hv;
  else               v[row * 512 + within] = hv;
}

// ---------------------------------------------------------------------------
// Kernel 2: sim[h,q,k] = (q_h @ k_h^T)/8 with sim[:,:,0] = -1e9.
// One wave per 16x16 tile: 8 heads * 32 * 32 tiles = 8192 waves.
// ---------------------------------------------------------------------------
__global__ __launch_bounds__(256) void k_sim(const _Float16* __restrict__ q,
                                             const _Float16* __restrict__ k,
                                             float* __restrict__ sim) {
  int wave = (blockIdx.x * blockDim.x + threadIdx.x) >> 5;
  int lane = threadIdx.x & 31;
  int h  = wave >> 10;         // 1024 tiles per head
  int qt = (wave >> 5) & 31;
  int kt = wave & 31;
  int m  = lane & 15;
  int hi = lane >> 4;

  const _Float16* qrow = q + (qt * 16 + m) * 512 + h * 64;  // row m of A, contiguous D
  const _Float16* kcol = k + (kt * 16 + m) * 512 + h * 64;  // column m of B

  v8f c = {};
  #pragma unroll
  for (int ks = 0; ks < 2; ++ks) {       // D = 64 = 2 * 32
    v16h a, b;
    int c0 = ks * 32 + hi * 16;
    #pragma unroll
    for (int e = 0; e < 8; ++e) { a[e] = qrow[c0 + e]; a[e + 8] = qrow[c0 + 8 + e]; }
    int kb = ks * 32 + hi * 16;
    #pragma unroll
    for (int e = 0; e < 16; ++e) b[e] = kcol[kb + e];
    c = wmma16(a, b, c);
  }
  int col = kt * 16 + m;
  #pragma unroll
  for (int r = 0; r < 8; ++r) {
    int row = qt * 16 + r + hi * 8;
    float val = (col == 0) ? -1e9f : c[r] * 0.125f;  // 1/sqrt(64)
    sim[h * 512 * 512 + row * 512 + col] = val;
  }
}

// ---------------------------------------------------------------------------
// Kernel 3: per batch row b (4096 blocks, 8 waves):
//   stage packed weights via GLOBAL_LOAD_ASYNC_TO_LDS_B128 (ASYNCcnt path),
//   wave = head: gather sim -> softmax -> attn @ V (WMMA) -> LayerNorm -> xln,
//   then all waves: 3-layer MLP over 1024 (l,d) rows via WMMA, masked pooling.
// ---------------------------------------------------------------------------
__global__ __launch_bounds__(256) void k_attn_mlp(
    const int* __restrict__ inputs, const float* __restrict__ sim,
    const _Float16* __restrict__ vmat, const unsigned char* __restrict__ blob,
    float* __restrict__ word_ft) {
  __shared__ int tok[16];
  __shared__ _Float16 xln[1024][8];                         // 16KB
  __shared__ __align__(16) unsigned char wblob[BLOB_BYTES]; // 16KB packed weights
  __shared__ float wordAcc[512];                            // 2KB
  __shared__ __align__(16) unsigned char wscratch[8][8192]; // 64KB per-wave scratch

  int b = blockIdx.x;
  int tid = threadIdx.x;
  int wave = tid >> 5;
  int lane = tid & 31;
  int m = lane & 15, hi = lane >> 4;

  // ---- stage weight blob: async global->LDS, 512B per instruction per wave ----
  {
    // LDS aperture: flat_addr[31:0] == LDS byte offset (ISA 10.2 aperture table)
    unsigned lds_base = (unsigned)(size_t)(&wblob[0]);
    #pragma unroll
    for (int it = 0; it < 4; ++it) {                 // 8 waves * 4 * 512B = 16KB
      int off = (wave * 4 + it) * 512 + lane * 16;
      unsigned ldsa = lds_base + (unsigned)off;
      unsigned long long ga = (unsigned long long)(blob + off);
      asm volatile("global_load_async_to_lds_b128 %0, %1, off"
                   :: "v"(ldsa), "v"(ga) : "memory");
    }
    asm volatile("s_wait_asynccnt 0x0" ::: "memory");
  }
  if (tid < 16) tok[tid] = inputs[b * 16 + tid];
  for (int i = tid; i < 512; i += 256) wordAcc[i] = 0.f;
  __syncthreads();

  const _Float16* w1T = (const _Float16*)(wblob);           // [64][32]
  const _Float16* w2T = (const _Float16*)(wblob + 4096);    // [64][64]
  const _Float16* w3T = (const _Float16*)(wblob + 12288);   // [16][64]
  const float*    fb  = (const float*)(wblob + 14336);      // b1|b2|b3|gam|bet

  // ---------------- Phase B: attention, one wave per head ----------------
  {
    int h = wave;
    float*    otile = (float*)&wscratch[wave][0];        // 16x64 f32
    _Float16* ap    = (_Float16*)&wscratch[wave][4096];  // 16x16 f16 probs

    if (lane < 16) {
      const float* srow = sim + h * 512 * 512 + tok[lane] * 512;
      __builtin_prefetch(srow, 0, 1);
      float prow[16];
      float mx = -1e30f;
      #pragma unroll
      for (int j = 0; j < 16; ++j) { float s = srow[tok[j]]; prow[j] = s; mx = fmaxf(mx, s); }
      float sum = 0.f;
      #pragma unroll
      for (int j = 0; j < 16; ++j) { float e = __expf(prow[j] - mx); prow[j] = e; sum += e; }
      float inv = 1.f / sum;
      #pragma unroll
      for (int j = 0; j < 16; ++j) ap[lane * 16 + j] = (_Float16)(prow[j] * inv);
    }
    // A fragment (16x16 attn, K padded 16->32 with zeros); same-wave LDS order is safe.
    v16h a;
    #pragma unroll
    for (int e = 0; e < 8; ++e) { a[e] = hi ? (_Float16)0.f : ap[m * 16 + e]; a[e + 8] = hi ? (_Float16)0.f : ap[m * 16 + 8 + e]; }
    #pragma unroll
    for (int cch = 0; cch < 4; ++cch) {   // D = 4 * 16 columns
      v16h bf;
      if (hi == 0) {
        #pragma unroll
        for (int j = 0; j < 16; ++j) bf[j] = vmat[tok[j] * 512 + h * 64 + cch * 16 + m];
      } else {
        #pragma unroll
        for (int j = 0; j < 16; ++j) bf[j] = (_Float16)0.f;  // K=16..31 (zero attn)
      }
      v8f c = {};
      c = wmma16(a, bf, c);
      #pragma unroll
      for (int r = 0; r < 8; ++r) otile[(r + hi * 8) * 64 + cch * 16 + m] = c[r];
    }
    // LayerNorm over d per row, write transposed (l,d,h) f16
    if (lane < 16) {
      float mu = 0.f;
      for (int d = 0; d < 64; ++d) mu += otile[lane * 64 + d];
      mu *= (1.f / 64.f);
      float var = 0.f;
      for (int d = 0; d < 64; ++d) { float t = otile[lane * 64 + d] - mu; var += t * t; }
      var *= (1.f / 64.f);
      float rstd = rsqrtf(var + 1e-6f);
      for (int d = 0; d < 64; ++d) {
        float y = (otile[lane * 64 + d] - mu) * rstd * fb[144 + d] + fb[208 + d];
        xln[lane * 64 + d][h] = (_Float16)y;
      }
    }
  }
  __syncthreads();

  // ---------------- Phase C: MLP over 1024 rows, 16-row chunks ----------------
  {
    _Float16* h1 = (_Float16*)&wscratch[wave][0];     // 16x64 f16
    _Float16* h2 = (_Float16*)&wscratch[wave][2048];  // 16x64 f16
    for (int chunk = wave; chunk < 64; chunk += 8) {
      int g0 = chunk * 16;
      // Layer 1: A = xln rows (K=8, pad to 32)
      v16h a1;
      if (hi == 0) {
        #pragma unroll
        for (int e = 0; e < 8; ++e) a1[e] = xln[g0 + m][e];
        #pragma unroll
        for (int e = 8; e < 16; ++e) a1[e] = (_Float16)0.f;
      } else {
        #pragma unroll
        for (int e = 0; e < 16; ++e) a1[e] = (_Float16)0.f;
      }
      #pragma unroll
      for (int nt = 0; nt < 4; ++nt) {
        v16h bf;
        #pragma unroll
        for (int e = 0; e < 16; ++e) bf[e] = w1T[(nt * 16 + m) * 32 + hi * 16 + e];
        v8f c = {};
        c = wmma16(a1, bf, c);
        int n = nt * 16 + m;
        #pragma unroll
        for (int r = 0; r < 8; ++r)
          h1[(r + hi * 8) * 64 + n] = (_Float16)fmaxf(c[r] + fb[n], 0.f);
      }
      // Layer 2: K = 64 (two k-steps)
      #pragma unroll
      for (int nt = 0; nt < 4; ++nt) {
        v8f c = {};
        #pragma unroll
        for (int ks = 0; ks < 2; ++ks) {
          v16h a2, bf;
          int c0 = ks * 32 + hi * 16;
          #pragma unroll
          for (int e = 0; e < 8; ++e) { a2[e] = h1[m * 64 + c0 + e]; a2[e + 8] = h1[m * 64 + c0 + 8 + e]; }
          int kb = ks * 32 + hi * 16;
          #pragma unroll
          for (int e = 0; e < 16; ++e) bf[e] = w2T[(nt * 16 + m) * 64 + kb + e];
          c = wmma16(a2, bf, c);
        }
        int n = nt * 16 + m;
        #pragma unroll
        for (int r = 0; r < 8; ++r)
          h2[(r + hi * 8) * 64 + n] = (_Float16)fmaxf(c[r] + fb[64 + n], 0.f);
      }
      // Layer 3: N = 8 (padded to 16), K = 64
      {
        v8f c = {};
        #pragma unroll
        for (int ks = 0; ks < 2; ++ks) {
          v16h a3, bf;
          int c0 = ks * 32 + hi * 16;
          #pragma unroll
          for (int e = 0; e < 8; ++e) { a3[e] = h2[m * 64 + c0 + e]; a3[e + 8] = h2[m * 64 + c0 + 8 + e]; }
          int kb = ks * 32 + hi * 16;
          #pragma unroll
          for (int e = 0; e < 16; ++e) bf[e] = w3T[m * 64 + kb + e];
          c = wmma16(a3, bf, c);
        }
        if (m < 8) {  // valid output features
          #pragma unroll
          for (int r = 0; r < 8; ++r) {
            int g = g0 + r + hi * 8;
            int l = g >> 6, d = g & 63;
            if (tok[l] != 0)
              atomicAdd(&wordAcc[d * 8 + m], c[r] + fb[128 + m]);
          }
        }
      }
    }
  }
  __syncthreads();

  int cnt = 0;
  for (int l = 0; l < 16; ++l) cnt += (tok[l] != 0) ? 1 : 0;
  float inv = 1.f / (float)cnt;
  for (int i = tid; i < 512; i += 256) word_ft[b * 512 + i] = wordAcc[i] * inv;
}

// ---------------------------------------------------------------------------
// Kernel 4: name_ft = mean over 8 words. out is (512 names, 512 feats) flat.
// ---------------------------------------------------------------------------
__global__ void k_name(const float* __restrict__ word_ft, float* __restrict__ out) {
  int idx = blockIdx.x * blockDim.x + threadIdx.x;
  if (idx >= 512 * 512) return;
  int nm = idx >> 9;
  int f = idx & 511;
  float s = 0.f;
  #pragma unroll
  for (int w = 0; w < 8; ++w) s += word_ft[(nm * 8 + w) * 512 + f];
  out[idx] = s * 0.125f;
}

extern "C" void kernel_launch(void* const* d_in, const int* in_sizes, int n_in,
                              void* d_out, int out_size, void* d_ws, size_t ws_size,
                              hipStream_t stream) {
  const int*   inputs = (const int*)  d_in[0];
  const float* emb    = (const float*)d_in[1];
  const float* w_qkv  = (const float*)d_in[2];
  const float* gamma  = (const float*)d_in[3];
  const float* beta   = (const float*)d_in[4];
  const float* w1     = (const float*)d_in[5];
  const float* b1     = (const float*)d_in[6];
  const float* w2     = (const float*)d_in[7];
  const float* b2     = (const float*)d_in[8];
  const float* w3     = (const float*)d_in[9];
  const float* b3     = (const float*)d_in[10];

  char* ws = (char*)d_ws;
  _Float16* qh = (_Float16*)(ws);                        // 512 KB
  _Float16* kh = (_Float16*)(ws + (size_t)512 * 1024);   // 512 KB
  _Float16* vh = (_Float16*)(ws + (size_t)1024 * 1024);  // 512 KB
  float* sim    = (float*)(ws + (size_t)1536 * 1024);    // 8 MB
  float* wordft = (float*)(ws + (size_t)(1536 + 8192) * 1024);      // 8 MB
  unsigned char* blob = (unsigned char*)(ws + (size_t)(1536 + 16384) * 1024);  // 16 KB
  size_t need = (size_t)(1536 + 16384) * 1024 + BLOB_BYTES;
  if (ws_size < need) return;

  k_prep<<<1, 256, 0, stream>>>(w1, w2, w3, b1, b2, b3, gamma, beta, blob);
  k_qkv<<<3072, 256, 0, stream>>>(emb, w_qkv, qh, kh, vh);
  k_sim<<<1024, 256, 0, stream>>>(qh, kh, sim);
  k_attn_mlp<<<4096, 256, 0, stream>>>(inputs, sim, vh, blob, wordft);
  k_name<<<1024, 256, 0, stream>>>(wordft, (float*)d_out);
}
